// MixtralBlockSparseTop2MLP_85435489452374
// MI455X (gfx1250) — matrix operand order repaired and checked by reading, not compiled, not run on hardware
//
#include <hip/hip_runtime.h>

#define NE 8
#define NR 4096
#define NH 1024
#define NF 3584
#define TWO_F (2 * NF)
#define BM 128
#define BK 32
#define SA 40          // f16 LDS row stride in halves (32 + 8 pad)
#define SAF 36         // f32 LDS row stride in floats (32 + 4 pad)
#define MAXT 40        // max M-tiles: R/BM + E

typedef __attribute__((ext_vector_type(16))) _Float16 v16h;
typedef __attribute__((ext_vector_type(2)))  __fp16   v2fp16;  // cvt_pkrtz result type
typedef __attribute__((ext_vector_type(8)))  float    v8f;

union FragH { v16h h; uint4 u[2]; v2fp16 p[8]; };

// ---------------------------------------------------------------------------
// Build per-expert M-tile table from device-side row counts.
// ---------------------------------------------------------------------------
__global__ void moe_prepare(const int* __restrict__ rows, int4* __restrict__ table) {
  if (threadIdx.x == 0) {
    int off = 0, n = 0;
    for (int e = 0; e < NE; ++e) {
      int c = rows[e];
      for (int t = 0; t < c; t += BM) {
        int v = c - t; if (v > BM) v = BM;
        table[n++] = make_int4(e, off + t, v, 0);
      }
      off += c;
    }
    for (; n < MAXT; ++n) table[n] = make_int4(0, 0, 0, 0);
  }
}

// ---------------------------------------------------------------------------
// GEMM1: x[rows,H] (f32) @ w1[e][H,2F] -> silu(gate)*up -> act (f16)
// Wave = 32 rows x 64 cols (2 A frags x 4 B frags). LDS W-tile interleaves
// gate/up per 16-col N-tile so the silu*up pair stays wave-local.
// ---------------------------------------------------------------------------
__global__ __launch_bounds__(256, 2) void moe_gemm1(
    const float* __restrict__ hs, const float* __restrict__ w1,
    const int4* __restrict__ table, _Float16* __restrict__ act)
{
  __shared__ __align__(16) float     sAf[2][BM * SAF];  // 128 rows x 32 k (f32)
  __shared__ __align__(16) _Float16  sW [2][BM * SA];   // 8 N-tiles x 16 cols x 32 k

  const int4 ti = table[blockIdx.y];
  const int rowsValid = ti.z;
  if (rowsValid <= 0) return;
  const int rowStart = ti.y;
  const float* __restrict__ w1e = w1 + (size_t)ti.x * NH * TWO_F;
  const int c0 = blockIdx.x * 64;

  const int tid  = threadIdx.x;
  const int lane = tid & 31;
  const int wave = tid >> 5;
  const int ln   = lane & 15;
  const int hi   = lane >> 4;
  const int mg   = wave >> 1;      // M-group: rows [mg*32, mg*32+32)
  const int ng   = wave & 1;       // N-group: LDS tiles [ng*4, ng*4+4)

  v8f acc[8];                      // acc[mi*4 + ni]
  #pragma unroll
  for (int i = 0; i < 8; ++i) acc[i] = v8f{};

  // --- async stage A: 128 rows x 128B, 4 x b128 per thread -----------------
  auto stageA = [&](int buf, int k0) {
    #pragma unroll
    for (int i = 0; i < 4; ++i) {
      const int id  = tid + 256 * i;       // 0..1023
      const int r   = id >> 3;
      const int seg = id & 7;              // 8 x 16B segments per row
      int gr = rowStart + r;
      if (gr >= NR) gr = 0;                // clamp tail (data discarded)
      const unsigned long long src =
          (unsigned long long)(const void*)(hs + (size_t)gr * NH + k0 + seg * 4);
      const unsigned int dst =
          (unsigned int)(unsigned long long)(const void*)(&sAf[buf][r * SAF + seg * 4]);
      asm volatile("global_load_async_to_lds_b128 %0, %1, off"
                   :: "v"(dst), "v"(src) : "memory");
    }
  };

  // --- stage W: one column per thread, 16 strided b32 loads (coalesced),
  //     packed-cvt, two b128 DS stores. Even tiles gate, odd tiles up. -----
  auto stageW = [&](int buf, int k0) {
    const int c  = tid & 127;
    const int kb = (tid >> 7) * 16;
    const int t  = c >> 4;                         // N-tile 0..7
    const int nn = c & 15;
    const int base = c0 + (t >> 1) * 16 + nn;
    const int gc = (t & 1) ? (NF + base) : base;   // odd tile -> up column
    const float* __restrict__ src = w1e + (size_t)(k0 + kb) * TWO_F + gc;
    float v[16];
    #pragma unroll
    for (int i = 0; i < 16; ++i) v[i] = src[(size_t)i * TWO_F];
    FragH wf;
    #pragma unroll
    for (int i = 0; i < 8; ++i)
      wf.p[i] = __builtin_amdgcn_cvt_pkrtz(v[2 * i], v[2 * i + 1]);
    _Float16* dst = &sW[buf][c * SA + kb];
    *(uint4*)(dst)     = wf.u[0];
    *(uint4*)(dst + 8) = wf.u[1];
  };

  stageA(0, 0);
  stageW(0, 0);
  int buf = 0;
  for (int kt = 0; kt < NH / BK; ++kt) {
    asm volatile("s_wait_asynccnt 0x0" ::: "memory");
    __syncthreads();
    if (kt + 1 < NH / BK) {
      stageA(buf ^ 1, (kt + 1) * BK);
      stageW(buf ^ 1, (kt + 1) * BK);
    }

    // 2 A fragments (f32 -> pk-cvt -> f16 WMMA layout)
    FragH af[2];
    #pragma unroll
    for (int mi = 0; mi < 2; ++mi) {
      const float* ab = &sAf[buf][(mg * 32 + mi * 16 + ln) * SAF + hi * 8];
      float4 a0 = *(const float4*)(ab);
      float4 a1 = *(const float4*)(ab + 4);
      float4 a2 = *(const float4*)(ab + 16);
      float4 a3 = *(const float4*)(ab + 20);
      af[mi].p[0] = __builtin_amdgcn_cvt_pkrtz(a0.x, a0.y);
      af[mi].p[1] = __builtin_amdgcn_cvt_pkrtz(a0.z, a0.w);
      af[mi].p[2] = __builtin_amdgcn_cvt_pkrtz(a1.x, a1.y);
      af[mi].p[3] = __builtin_amdgcn_cvt_pkrtz(a1.z, a1.w);
      af[mi].p[4] = __builtin_amdgcn_cvt_pkrtz(a2.x, a2.y);
      af[mi].p[5] = __builtin_amdgcn_cvt_pkrtz(a2.z, a2.w);
      af[mi].p[6] = __builtin_amdgcn_cvt_pkrtz(a3.x, a3.y);
      af[mi].p[7] = __builtin_amdgcn_cvt_pkrtz(a3.z, a3.w);
    }
    // 4 B fragments, then the 8-WMMA chain
    FragH bf[4];
    #pragma unroll
    for (int ni = 0; ni < 4; ++ni) {
      const _Float16* bb = &sW[buf][((ng * 4 + ni) * 16 + ln) * SA + hi * 8];
      bf[ni].u[0] = *(const uint4*)(bb);
      bf[ni].u[1] = *(const uint4*)(bb + 16);
    }
    #pragma unroll
    for (int ni = 0; ni < 4; ++ni)
      #pragma unroll
      for (int mi = 0; mi < 2; ++mi)
        acc[mi * 4 + ni] = __builtin_amdgcn_wmma_f32_16x16x32_f16(
            false, af[mi].h, false, bf[ni].h, (short)0, acc[mi * 4 + ni], false, false);
    buf ^= 1;
  }

  // Epilogue: silu(gate) * up (gate = even N-tile, up = odd N-tile)
  #pragma unroll
  for (int mi = 0; mi < 2; ++mi) {
    #pragma unroll
    for (int s = 0; s < 2; ++s) {
      const int coln = c0 + (2 * ng + s) * 16 + ln;
      #pragma unroll
      for (int j = 0; j < 8; ++j) {
        const int rr = mg * 32 + mi * 16 + hi * 8 + j;
        if (rr < rowsValid) {
          float g = acc[mi * 4 + 2 * s][j];
          float u = acc[mi * 4 + 2 * s + 1][j];
          float sg = g / (1.f + __expf(-g));
          act[(size_t)(rowStart + rr) * NF + coln] = (_Float16)(sg * u);
        }
      }
    }
  }
}

// ---------------------------------------------------------------------------
// GEMM2: act[rows,F] (f16) @ w2[e][F,H] (f32->f16) -> out (f32)
// ---------------------------------------------------------------------------
__global__ __launch_bounds__(256, 2) void moe_gemm2(
    const _Float16* __restrict__ act, const float* __restrict__ w2,
    const int4* __restrict__ table, float* __restrict__ out)
{
  __shared__ __align__(16) _Float16 sA[2][BM * SA];
  __shared__ __align__(16) _Float16 sW[2][BM * SA];

  const int4 ti = table[blockIdx.y];
  const int rowsValid = ti.z;
  if (rowsValid <= 0) return;
  const int rowStart = ti.y;
  const float* __restrict__ w2e = w2 + (size_t)ti.x * NF * NH;
  const int n0 = blockIdx.x * 128;

  const int tid  = threadIdx.x;
  const int lane = tid & 31;
  const int wave = tid >> 5;
  const int ln   = lane & 15;
  const int hi   = lane >> 4;
  const int mg   = wave >> 1;
  const int ng   = wave & 1;

  v8f acc[8];
  #pragma unroll
  for (int i = 0; i < 8; ++i) acc[i] = v8f{};

  auto stageA = [&](int buf, int k0) {
    // 128 rows x 64B each = 512 x b128 async copies, 2 per thread
    #pragma unroll
    for (int i = 0; i < 2; ++i) {
      const int id  = tid + 256 * i;
      const int r   = id >> 2;
      const int seg = id & 3;
      int gr = rowStart + r;
      if (gr >= NR) gr = 0;
      const unsigned long long src =
          (unsigned long long)(const void*)(act + (size_t)gr * NF + k0 + seg * 8);
      const unsigned int dst =
          (unsigned int)(unsigned long long)(const void*)(&sA[buf][r * SA + seg * 8]);
      asm volatile("global_load_async_to_lds_b128 %0, %1, off"
                   :: "v"(dst), "v"(src) : "memory");
    }
  };

  auto stageW = [&](int buf, int k0) {
    const int c  = tid & 127;
    const int kb = (tid >> 7) * 16;
    const float* __restrict__ src = w2e + (size_t)(k0 + kb) * NH + n0 + c;
    float v[16];
    #pragma unroll
    for (int i = 0; i < 16; ++i) v[i] = src[(size_t)i * NH];
    FragH wf;
    #pragma unroll
    for (int i = 0; i < 8; ++i)
      wf.p[i] = __builtin_amdgcn_cvt_pkrtz(v[2 * i], v[2 * i + 1]);
    _Float16* dst = &sW[buf][c * SA + kb];
    *(uint4*)(dst)     = wf.u[0];
    *(uint4*)(dst + 8) = wf.u[1];
  };

  stageA(0, 0);
  stageW(0, 0);
  int buf = 0;
  for (int kt = 0; kt < NF / BK; ++kt) {
    asm volatile("s_wait_asynccnt 0x0" ::: "memory");
    __syncthreads();
    if (kt + 1 < NF / BK) {
      stageA(buf ^ 1, (kt + 1) * BK);
      stageW(buf ^ 1, (kt + 1) * BK);
    }

    FragH af[2];
    #pragma unroll
    for (int mi = 0; mi < 2; ++mi) {
      const _Float16* ab = &sA[buf][(mg * 32 + mi * 16 + ln) * SA + hi * 8];
      af[mi].u[0] = *(const uint4*)(ab);
      af[mi].u[1] = *(const uint4*)(ab + 16);
    }
    FragH bf[4];
    #pragma unroll
    for (int ni = 0; ni < 4; ++ni) {
      const _Float16* bb = &sW[buf][((ng * 4 + ni) * 16 + ln) * SA + hi * 8];
      bf[ni].u[0] = *(const uint4*)(bb);
      bf[ni].u[1] = *(const uint4*)(bb + 16);
    }
    #pragma unroll
    for (int ni = 0; ni < 4; ++ni)
      #pragma unroll
      for (int mi = 0; mi < 2; ++mi)
        acc[mi * 4 + ni] = __builtin_amdgcn_wmma_f32_16x16x32_f16(
            false, af[mi].h, false, bf[ni].h, (short)0, acc[mi * 4 + ni], false, false);
    buf ^= 1;
  }

  #pragma unroll
  for (int mi = 0; mi < 2; ++mi) {
    #pragma unroll
    for (int ni = 0; ni < 4; ++ni) {
      const int coln = n0 + (ng * 4 + ni) * 16 + ln;
      #pragma unroll
      for (int j = 0; j < 8; ++j) {
        const int rr = mg * 32 + mi * 16 + hi * 8 + j;
        if (rr < rowsValid)
          out[(size_t)(rowStart + rr) * NH + coln] = acc[mi * 4 + ni][j];
      }
    }
  }
}

// ---------------------------------------------------------------------------
extern "C" void kernel_launch(void* const* d_in, const int* in_sizes, int n_in,
                              void* d_out, int out_size, void* d_ws, size_t ws_size,
                              hipStream_t stream) {
  (void)in_sizes; (void)n_in; (void)out_size; (void)ws_size;
  const float* hs   = (const float*)d_in[0];
  const float* w1   = (const float*)d_in[1];
  const float* w2   = (const float*)d_in[2];
  const int*   rows = (const int*)d_in[3];
  float* out = (float*)d_out;

  int4*     table = (int4*)d_ws;
  _Float16* act   = (_Float16*)((char*)d_ws + 1024);  // R*F f16 = 29.4 MB

  moe_prepare<<<1, 32, 0, stream>>>(rows, table);
  moe_gemm1<<<dim3(NF / 64, MAXT), 256, 0, stream>>>(hs, w1, table, act);
  moe_gemm2<<<dim3(NH / 128, MAXT), 256, 0, stream>>>(act, w2, table, out);
}
